// GQAAttention_71605694759531
// MI455X (gfx1250) — matrix operand order repaired and checked by reading, not compiled, not run on hardware
//
#include <hip/hip_runtime.h>
#include <hip/hip_bf16.h>

typedef __bf16 bf16;
typedef __attribute__((ext_vector_type(16))) bf16  v16bf;
typedef __attribute__((ext_vector_type(8)))  bf16  bf16x8;
typedef __attribute__((ext_vector_type(8)))  float v8f;

typedef __attribute__((ext_vector_type(4))) unsigned int u32x4;
typedef __attribute__((ext_vector_type(8))) int          i32x8;
typedef __attribute__((ext_vector_type(4))) int          i32x4;

union Frag {
    v16bf  v;
    bf16x8 h[2];
};

#define B_SZ   4
#define S_LEN  2048
#define DMODEL 2048
#define HDIM   512
#define NGRP   4

// Tensor Data Mover availability (this toolchain: 6-arg builtin,
// amdgpu-toolchain / clang-23 signature)
#if defined(__has_builtin)
#if __has_builtin(__builtin_amdgcn_tensor_load_to_lds) && \
    __has_builtin(__builtin_amdgcn_s_wait_tensorcnt)
#define USE_TDM 1
#endif
#endif

#if defined(USE_TDM)
// Build a D# for a 2D tile of 16-bit elements and issue TENSOR_LOAD_TO_LDS.
//   lds_off          : byte offset of destination in LDS
//   gptr             : global address of tile start
//   tile_w / tile_h  : tile dims in elements (w = contiguous dim0)
//   stride_elems     : row stride of dim1 in elements
__device__ __forceinline__ void tdm_load_2d(unsigned int lds_off, const void* gptr,
                                            unsigned int tile_w, unsigned int tile_h,
                                            unsigned int stride_elems)
{
    unsigned long long ga = (unsigned long long)(uintptr_t)gptr;
    u32x4 g0;
    g0[0] = 1u;                                              // count=1, user mode
    g0[1] = lds_off;                                         // lds_addr (bytes)
    g0[2] = (unsigned int)ga;                                // global_addr[31:0]
    g0[3] = (unsigned int)((ga >> 32) & 0x01FFFFFFu)         // global_addr[56:32]
            | (2u << 30);                                    // type = 2 ("image")
    i32x8 g1;
    g1[0] = (int)(1u << 16);                                 // wg_mask=0, data_size=2B
    g1[1] = (int)(tile_w << 16);                             // tensor_dim0 lo16
    g1[2] = (int)((tile_w >> 16) | (tile_h << 16));          // tensor_dim0 hi | tensor_dim1 lo
    g1[3] = (int)((tile_h >> 16) | (tile_w << 16));          // tensor_dim1 hi | tile_dim0
    g1[4] = (int)tile_h;                                     // tile_dim1, tile_dim2=0
    g1[5] = (int)stride_elems;                               // tensor_dim0_stride lo32
    g1[6] = 0;                                               // stride hi, dim1_stride lo
    g1[7] = 0;
    i32x4 z4;
    z4[0] = 0; z4[1] = 0; z4[2] = 0; z4[3] = 0;
    i32x8 z8;
    z8[0] = 0; z8[1] = 0; z8[2] = 0; z8[3] = 0;
    z8[4] = 0; z8[5] = 0; z8[6] = 0; z8[7] = 0;
    __builtin_amdgcn_tensor_load_to_lds(g0, g1, z4, z4, z8, 0);
}
#endif

// ---------------------------------------------------------------------------
// f32 -> bf16 conversion
// ---------------------------------------------------------------------------
__global__ void cvt_bf16(const float* __restrict__ in, bf16* __restrict__ out, int n) {
    int i = blockIdx.x * 256 + threadIdx.x;
    if (i < n) out[i] = (bf16)in[i];
}

// ---------------------------------------------------------------------------
// v[B,S,HD] -> vT[B,HD,S]
// ---------------------------------------------------------------------------
__global__ void transpose_v(const bf16* __restrict__ v, bf16* __restrict__ vt) {
    int idx = blockIdx.x * 256 + threadIdx.x;         // B*S*HD total
    int f = idx % HDIM;
    int s = (idx / HDIM) % S_LEN;
    int b = idx / (HDIM * S_LEN);
    vt[((size_t)b * HDIM + f) * S_LEN + s] = v[idx];
}

// ---------------------------------------------------------------------------
// Tiled bf16 GEMM with f32 accumulation via v_wmma_f32_16x16x32_bf16.
// Block tile 128x128, BK=32, 8 waves (256 threads), wave tile 32x64.
// A tile staged by the Tensor Data Mover (wave 0 issues, TENSORcnt wait),
// B tile staged manually transposed (lds_bt[n][k]) for contiguous-K frags.
// ---------------------------------------------------------------------------
#define LDB 48   // padded LDS row stride for B^T (96B, 16B-aligned rows)

template <bool F32OUT>
__global__ __launch_bounds__(256) void gemm_bf16(
    const bf16* __restrict__ A, const bf16* __restrict__ Bw,
    const float* __restrict__ bias, void* __restrict__ outp,
    int M, int N, int K)
{
    __shared__ bf16 lds_a[128 * 32];     // A tile, row-major [m][k], stride 32
    __shared__ bf16 lds_bt[128 * LDB];   // B tile transposed [n][k]

    const int tid  = threadIdx.x;
    const int w    = tid >> 5;
    const int lane = tid & 31;
    const int nI   = lane & 15;
    const int h    = lane >> 4;
    const int wm   = w >> 1;             // 0..3  (M direction)
    const int wn   = w & 1;              // 0..1  (N direction)

    const int m0 = blockIdx.y * 128;
    const int n0 = blockIdx.x * 128;

    v8f zero = {};
    v8f acc[2][4];
#pragma unroll
    for (int i = 0; i < 2; ++i)
#pragma unroll
        for (int j = 0; j < 4; ++j) acc[i][j] = zero;

#if !defined(USE_TDM)
    const int arow = tid >> 1, aseg = (tid & 1) * 16;   // 128 rows x 32 cols
#endif
    const int brow = tid >> 3, bseg = (tid & 7) * 16;   // 32 rows x 128 cols

    for (int k0 = 0; k0 < K; k0 += 32) {
        __syncthreads();
#if defined(USE_TDM)
        // ---- TDM: DMA the 128x32 A tile straight into LDS ----
        if (tid < 32) {
            tdm_load_2d((unsigned int)(uintptr_t)(void*)lds_a,
                        A + (size_t)m0 * K + k0,
                        32u, 128u, (unsigned int)K);
        }
#else
        {
            const bf16* ap = A + (size_t)(m0 + arow) * K + k0 + aseg;
            *(bf16x8*)&lds_a[arow * 32 + aseg]     = *(const bf16x8*)ap;
            *(bf16x8*)&lds_a[arow * 32 + aseg + 8] = *(const bf16x8*)(ap + 8);
        }
#endif
        // ---- stage B tile transposed -> lds_bt[n][k] ----
        {
            const bf16* bp = Bw + (size_t)(k0 + brow) * N + n0 + bseg;
            union { bf16x8 v[2]; bf16 e[16]; } tmp;
            tmp.v[0] = *(const bf16x8*)bp;
            tmp.v[1] = *(const bf16x8*)(bp + 8);
            if (k0 + 32 < K) __builtin_prefetch(bp + (size_t)32 * N, 0, 0);
#pragma unroll
            for (int i = 0; i < 16; ++i)
                lds_bt[(bseg + i) * LDB + brow] = tmp.e[i];
        }
#if defined(USE_TDM)
        if (tid < 32) __builtin_amdgcn_s_wait_tensorcnt(0);
#endif
        __syncthreads();

        Frag af[2], bfr[4];
#pragma unroll
        for (int i = 0; i < 2; ++i) {
            const bf16* p = &lds_a[(wm * 32 + i * 16 + nI) * 32];
            af[i].h[0] = *(const bf16x8*)(p + 8 * h);
            af[i].h[1] = *(const bf16x8*)(p + 16 + 8 * h);
        }
#pragma unroll
        for (int j = 0; j < 4; ++j) {
            const bf16* p = &lds_bt[(wn * 64 + j * 16 + nI) * LDB];
            bfr[j].h[0] = *(const bf16x8*)(p + 8 * h);
            bfr[j].h[1] = *(const bf16x8*)(p + 16 + 8 * h);
        }
#pragma unroll
        for (int i = 0; i < 2; ++i)
#pragma unroll
            for (int j = 0; j < 4; ++j)
                acc[i][j] = __builtin_amdgcn_wmma_f32_16x16x32_bf16(
                    false, af[i].v, false, bfr[j].v, (short)0, acc[i][j], false, false);
    }

#pragma unroll
    for (int i = 0; i < 2; ++i)
#pragma unroll
        for (int j = 0; j < 4; ++j) {
            int ng = n0 + wn * 64 + j * 16 + nI;
            float bv = bias[ng];
#pragma unroll
            for (int r = 0; r < 8; ++r) {
                int rg = m0 + wm * 32 + i * 16 + h * 8 + r;
                float val = acc[i][j][r] + bv;
                if (F32OUT) ((float*)outp)[(size_t)rg * N + ng] = val;
                else        ((bf16*)outp)[(size_t)rg * N + ng] = (bf16)val;
            }
        }
}

// ---------------------------------------------------------------------------
// Flash-style GQA attention.
// One block per (b, g, 16-query-row tile). 8 waves:
//   QK^T: wave w covers keys [kt*128 + w*16, +16); online softmax across
//   waves via LDS; P shared via LDS (bf16); PV: wave w covers output
//   features [w*64, w*64+64).
// ---------------------------------------------------------------------------
__global__ __launch_bounds__(256) void gqa_attn(
    const bf16* __restrict__ Q,   // [B, S, DMODEL]   (group g at cols g*HDIM)
    const bf16* __restrict__ K,   // [B, S, HDIM]
    const bf16* __restrict__ VT,  // [B, HDIM, S]
    bf16* __restrict__ O)         // [B, S, DMODEL]
{
    __shared__ float lds_max[8][16];
    __shared__ float lds_sum[8][16];
    __shared__ bf16  lds_p[16 * 128];

    const int bid = blockIdx.x;
    const int qt  = bid & 127;           // S/16 = 128 query tiles
    const int g   = (bid >> 7) & 3;
    const int b   = bid >> 9;

    const int tid  = threadIdx.x;
    const int w    = tid >> 5;
    const int lane = tid & 31;
    const int nI   = lane & 15;
    const int h    = lane >> 4;

    const float scale = 0.044194173824159216f;   // 1/sqrt(512)

    // Preload Q fragments for this wave's 16 query rows (16 K-chunks of 32)
    Frag qf[16];
    {
        const bf16* qp = Q + (size_t)(b * S_LEN + qt * 16 + nI) * DMODEL + g * HDIM;
#pragma unroll
        for (int fc = 0; fc < 16; ++fc) {
            qf[fc].h[0] = *(const bf16x8*)(qp + fc * 32 + 8 * h);
            qf[fc].h[1] = *(const bf16x8*)(qp + fc * 32 + 16 + 8 * h);
        }
    }

    v8f zero = {};
    v8f oacc[4];
#pragma unroll
    for (int t = 0; t < 4; ++t) oacc[t] = zero;
    float m_run[8], l_run[8];
#pragma unroll
    for (int r = 0; r < 8; ++r) { m_run[r] = -1e30f; l_run[r] = 0.0f; }

    const bf16* kb  = K  + (size_t)b * S_LEN * HDIM;
    const bf16* vtb = VT + (size_t)b * HDIM * S_LEN;

    for (int kt = 0; kt < S_LEN / 128; ++kt) {
        // ---- scores: wave's 16x16 tile of QK^T ----
        v8f s = zero;
        {
            const int key = kt * 128 + w * 16 + nI;
            const bf16* kp = kb + (size_t)key * HDIM;
#pragma unroll
            for (int fc = 0; fc < 16; ++fc) {
                Frag kf;
                kf.h[0] = *(const bf16x8*)(kp + fc * 32 + 8 * h);
                kf.h[1] = *(const bf16x8*)(kp + fc * 32 + 16 + 8 * h);
                s = __builtin_amdgcn_wmma_f32_16x16x32_bf16(
                    false, qf[fc].v, false, kf.v, (short)0, s, false, false);
            }
        }
#pragma unroll
        for (int r = 0; r < 8; ++r) s[r] *= scale;

        // ---- row max within wave (16 key cols), then across waves ----
        float rmax[8];
#pragma unroll
        for (int r = 0; r < 8; ++r) {
            float m = s[r];
            m = fmaxf(m, __shfl_xor(m, 1, 16));
            m = fmaxf(m, __shfl_xor(m, 2, 16));
            m = fmaxf(m, __shfl_xor(m, 4, 16));
            m = fmaxf(m, __shfl_xor(m, 8, 16));
            rmax[r] = m;
        }
        if (nI == 0) {
#pragma unroll
            for (int r = 0; r < 8; ++r) lds_max[w][h * 8 + r] = rmax[r];
        }
        __syncthreads();

        float mnew[8], alpha[8];
#pragma unroll
        for (int r = 0; r < 8; ++r) {
            float m = m_run[r];
#pragma unroll
            for (int wv = 0; wv < 8; ++wv) m = fmaxf(m, lds_max[wv][h * 8 + r]);
            mnew[r]  = m;
            alpha[r] = __expf(m_run[r] - m);
        }

        // ---- p = exp(s - mnew), row sums, share P tile ----
        float rsum[8];
#pragma unroll
        for (int r = 0; r < 8; ++r) {
            float p = __expf(s[r] - mnew[r]);
            s[r] = p;
            float t2 = p;
            t2 += __shfl_xor(t2, 1, 16);
            t2 += __shfl_xor(t2, 2, 16);
            t2 += __shfl_xor(t2, 4, 16);
            t2 += __shfl_xor(t2, 8, 16);
            rsum[r] = t2;
            lds_p[(h * 8 + r) * 128 + w * 16 + nI] = (bf16)p;
        }
        if (nI == 0) {
#pragma unroll
            for (int r = 0; r < 8; ++r) lds_sum[w][h * 8 + r] = rsum[r];
        }
        __syncthreads();

#pragma unroll
        for (int r = 0; r < 8; ++r) {
            float tsum = 0.0f;
#pragma unroll
            for (int wv = 0; wv < 8; ++wv) tsum += lds_sum[wv][h * 8 + r];
            l_run[r] = alpha[r] * l_run[r] + tsum;
            m_run[r] = mnew[r];
        }
#pragma unroll
        for (int t = 0; t < 4; ++t)
#pragma unroll
            for (int r = 0; r < 8; ++r) oacc[t][r] *= alpha[r];

        // ---- PV: P (16x128, LDS) x V (128 x 64-per-wave, from vT) ----
#pragma unroll
        for (int c = 0; c < 4; ++c) {
            Frag pf;
            const bf16* pp = &lds_p[nI * 128 + c * 32];
            pf.h[0] = *(const bf16x8*)(pp + 8 * h);
            pf.h[1] = *(const bf16x8*)(pp + 16 + 8 * h);
#pragma unroll
            for (int t = 0; t < 4; ++t) {
                const int feat = w * 64 + t * 16 + nI;
                const bf16* vp = vtb + (size_t)feat * S_LEN + kt * 128 + c * 32;
                Frag vf;
                vf.h[0] = *(const bf16x8*)(vp + 8 * h);
                vf.h[1] = *(const bf16x8*)(vp + 16 + 8 * h);
                oacc[t] = __builtin_amdgcn_wmma_f32_16x16x32_bf16(
                    false, pf.v, false, vf.v, (short)0, oacc[t], false, false);
            }
        }
        __syncthreads();
    }

    // ---- normalize and store O (bf16) ----
#pragma unroll
    for (int t = 0; t < 4; ++t) {
        const int col = g * HDIM + w * 64 + t * 16 + nI;
#pragma unroll
        for (int r = 0; r < 8; ++r) {
            const int row = qt * 16 + h * 8 + r;
            float val = oacc[t][r] / l_run[r];
            O[(size_t)(b * S_LEN + row) * DMODEL + col] = (bf16)val;
        }
    }
}

// ---------------------------------------------------------------------------
// launcher
// ---------------------------------------------------------------------------
extern "C" void kernel_launch(void* const* d_in, const int* in_sizes, int n_in,
                              void* d_out, int out_size, void* d_ws, size_t ws_size,
                              hipStream_t stream) {
    (void)in_sizes; (void)n_in; (void)out_size; (void)ws_size;

    const float* x  = (const float*)d_in[0];
    const float* Wq = (const float*)d_in[1];
    const float* bq = (const float*)d_in[2];
    const float* Wk = (const float*)d_in[3];
    const float* bk = (const float*)d_in[4];
    const float* Wv = (const float*)d_in[5];
    const float* bv = (const float*)d_in[6];
    const float* Wo = (const float*)d_in[7];
    const float* bo = (const float*)d_in[8];
    float* out = (float*)d_out;

    const size_t M = (size_t)B_SZ * S_LEN;   // 8192

    char* ws = (char*)d_ws;
    size_t off = 0;
    auto take = [&](size_t bytes) -> char* {
        char* p = ws + off;
        off = (off + bytes + 255) & ~(size_t)255;
        return p;
    };
    bf16* xb  = (bf16*)take(M * DMODEL * 2);
    bf16* wqb = (bf16*)take((size_t)DMODEL * DMODEL * 2);
    bf16* wkb = (bf16*)take((size_t)DMODEL * HDIM * 2);
    bf16* wvb = (bf16*)take((size_t)DMODEL * HDIM * 2);
    bf16* wob = (bf16*)take((size_t)DMODEL * DMODEL * 2);
    bf16* qb  = (bf16*)take(M * DMODEL * 2);
    bf16* kb  = (bf16*)take(M * HDIM * 2);
    bf16* vb  = (bf16*)take(M * HDIM * 2);
    bf16* vtb = (bf16*)take(M * HDIM * 2);
    bf16* ob  = (bf16*)take(M * DMODEL * 2);

    // 1) fp32 -> bf16 conversions
    {
        int n;
        n = (int)(M * DMODEL);   cvt_bf16<<<n / 256, 256, 0, stream>>>(x,  xb,  n);
        n = DMODEL * DMODEL;     cvt_bf16<<<n / 256, 256, 0, stream>>>(Wq, wqb, n);
        n = DMODEL * HDIM;       cvt_bf16<<<n / 256, 256, 0, stream>>>(Wk, wkb, n);
        n = DMODEL * HDIM;       cvt_bf16<<<n / 256, 256, 0, stream>>>(Wv, wvb, n);
        n = DMODEL * DMODEL;     cvt_bf16<<<n / 256, 256, 0, stream>>>(Wo, wob, n);
    }

    dim3 blk(256);
    // 2) projections (bf16 out)
    gemm_bf16<false><<<dim3(DMODEL / 128, (unsigned)(M / 128)), blk, 0, stream>>>(
        xb, wqb, bq, (void*)qb, (int)M, DMODEL, DMODEL);
    gemm_bf16<false><<<dim3(HDIM / 128, (unsigned)(M / 128)), blk, 0, stream>>>(
        xb, wkb, bk, (void*)kb, (int)M, HDIM, DMODEL);
    gemm_bf16<false><<<dim3(HDIM / 128, (unsigned)(M / 128)), blk, 0, stream>>>(
        xb, wvb, bv, (void*)vb, (int)M, HDIM, DMODEL);

    // 3) v -> vT
    {
        int n = (int)(M * HDIM);
        transpose_v<<<n / 256, 256, 0, stream>>>(vb, vtb);
    }

    // 4) attention (flash-style, streaming softmax)
    gqa_attn<<<B_SZ * NGRP * (S_LEN / 16), blk, 0, stream>>>(qb, kb, vtb, ob);

    // 5) output projection -> f32 d_out
    gemm_bf16<true><<<dim3(DMODEL / 128, (unsigned)(M / 128)), blk, 0, stream>>>(
        ob, wob, bo, (void*)out, (int)M, DMODEL, DMODEL);
}
